// LinearAttention_9397388444314
// MI455X (gfx1250) — matrix operand order repaired and checked by reading, not compiled, not run on hardware
//
#include <hip/hip_runtime.h>
#include <hip/hip_bf16.h>

typedef __bf16 bf16;
typedef __attribute__((ext_vector_type(16))) __bf16 v16bf;
typedef __attribute__((ext_vector_type(8)))  float   v8f;
typedef __attribute__((ext_vector_type(4)))  int     v4i_;

#define DH    1024
#define SEQ   4096
#define MROWS 32768   /* N * S = 8 * 4096 */

/* ---- workspace layout (bytes) ---- */
#define QB_OFF   (0ull)                        /* Q bf16:  64 MB                */
#define XB_OFF   (67108864ull)                 /* x bf16:  64 MB                */
#define WQT_OFF  (134217728ull)                /* Wq^T bf16: 2 MB               */
#define WKT_OFF  (WQT_OFF + 2097152ull)
#define WVT_OFF  (WQT_OFF + 4194304ull)
#define WOT_OFF  (WQT_OFF + 6291456ull)
#define KV_OFF   (WQT_OFF + 8388608ull)        /* KV f32: 32 KB                 */
#define KS_OFF   (KV_OFF + 32768ull)           /* Ksum f32: 32 KB               */

/* ---- CDNA5 async global->LDS copy (ASYNCcnt) with safe fallback ----------
 * Probe-confirmed signature: (v4i AS1* gsrc, v4i AS3* lds, Imm offset, Imm cpol)
 */
#if defined(__gfx1250__) && __has_builtin(__builtin_amdgcn_global_load_async_to_lds_b128)
#define HAVE_ASYNC 1
#define ASYNC_CP16(ldsp, gp)                                                   \
  __builtin_amdgcn_global_load_async_to_lds_b128(                              \
      (__attribute__((address_space(1))) v4i_*)(uintptr_t)(gp),                \
      (__attribute__((address_space(3))) v4i_*)(uint32_t)(uintptr_t)(ldsp),    \
      0, 0)
#else
#define ASYNC_CP16(ldsp, gp) (*(uint4*)(ldsp) = *(const uint4*)(gp))
#endif

#if defined(__gfx1250__) && __has_builtin(__builtin_amdgcn_s_wait_asynccnt)
#define WAIT_ASYNC(n) __builtin_amdgcn_s_wait_asynccnt(n)
#elif defined(HAVE_ASYNC)
#define WAIT_ASYNC(n) asm volatile("s_wait_asynccnt %0" ::"i"(n) : "memory")
#else
#define WAIT_ASYNC(n) ((void)0)
#endif

union FragU {            /* 16 bf16 = 32B, loaded as two b128 LDS reads */
  uint4 q[2];
  v16bf v;
};

__device__ __forceinline__ float elup1(float t) {
  return t > 0.f ? t + 1.f : __expf(t);   /* elu(t)+1 */
}

/* ---- fp32 -> bf16 elementwise (for x) ------------------------------------ */
__global__ void cvt_bf16_kernel(const float* __restrict__ src,
                                bf16* __restrict__ dst, int n) {
  int i = blockIdx.x * blockDim.x + threadIdx.x;
  if (i < n) dst[i] = (bf16)src[i];
}

/* ---- fp32 -> bf16 32x32 tiled TRANSPOSE (weights -> W^T[n][k]) ----------- */
__global__ __launch_bounds__(256)
void cvtT_kernel(const float* __restrict__ src, bf16* __restrict__ dst) {
  __shared__ float t[32][33];
  const int bn = blockIdx.x * 32;   /* n base (src cols)  */
  const int bk = blockIdx.y * 32;   /* k base (src rows)  */
  const int tx = threadIdx.x & 31;
  const int ty = threadIdx.x >> 5;  /* 0..7 */
  #pragma unroll
  for (int i = 0; i < 32; i += 8)
    t[ty + i][tx] = src[(size_t)(bk + ty + i) * DH + bn + tx];
  __syncthreads();
  #pragma unroll
  for (int i = 0; i < 32; i += 8)
    dst[(size_t)(bn + ty + i) * DH + bk + tx] = (bf16)t[tx][ty + i];
}

/* ---------------- Pass A: fused Q/K/V projection + sequence reductions -----
 * Block tile 128(M) x 64(N), 8 waves as 4(M) x 2(N); wave tile 32x32 per
 * matrix -> 12 WMMAs per k-step per wave from 2 shared A fragments.
 * Double-buffered LDS stages filled with async global->LDS b128 copies.
 */
__global__ __launch_bounds__(256)
void qkv_kernel(const bf16* __restrict__ xb,
                const bf16* __restrict__ WqT, const bf16* __restrict__ WkT,
                const bf16* __restrict__ WvT,
                const float* __restrict__ bq, const float* __restrict__ bk,
                const float* __restrict__ bv,
                bf16* __restrict__ Qb,
                float* __restrict__ KV, float* __restrict__ Ksum) {
  __shared__ bf16 As[2][128][40];    /* x tile  (row-major, 80B padded rows) */
  __shared__ bf16 Bq[2][64][40];     /* W^T tiles: [n][k], verbatim copies   */
  __shared__ bf16 Bk[2][64][40];
  __shared__ bf16 Bv[2][64][40];
  __shared__ float redKV[64];
  __shared__ float redKS[64];

  const int tid  = threadIdx.x;
  const int lane = tid & 31;
  const int w    = tid >> 5;
  const int wm   = w & 3;            /* wave M 0..3 */
  const int wn   = w >> 2;           /* wave N 0..1 */
  const int l15  = lane & 15;
  const int hi   = lane >> 4;

  const int gmBase = blockIdx.y * 128;
  const int gnBase = blockIdx.x * 64;
  const int nb     = gmBase / SEQ;   /* batch (128 | 4096) */

  if (tid < 64) { redKV[tid] = 0.f; redKS[tid] = 0.f; }

  v8f acc[3][2][2] = {};

  auto stage = [&](int k0, int buf) {
    /* A: 128 rows x 32 k = 512 x 16B chunks, 2 per thread */
    #pragma unroll
    for (int t = 0; t < 2; ++t) {
      int ch = tid + 256 * t;
      int r = ch >> 2, kc = (ch & 3) * 8;
      ASYNC_CP16(&As[buf][r][kc], xb + (size_t)(gmBase + r) * DH + k0 + kc);
    }
    /* B: per matrix 64 rows x 32 k = 256 chunks, 1 per thread */
    int n = tid >> 2, kc = (tid & 3) * 8;
    size_t gi = (size_t)(gnBase + n) * DH + k0 + kc;
    ASYNC_CP16(&Bq[buf][n][kc], WqT + gi);
    ASYNC_CP16(&Bk[buf][n][kc], WkT + gi);
    ASYNC_CP16(&Bv[buf][n][kc], WvT + gi);
  };

  stage(0, 0);
  const int NK = DH / 32;
  for (int ks = 0; ks < NK; ++ks) {
    const int buf = ks & 1;
    if (ks + 1 < NK) { stage((ks + 1) * 32, buf ^ 1); WAIT_ASYNC(5); }
    else             { WAIT_ASYNC(0); }
    __syncthreads();

    FragU a[2];
    #pragma unroll
    for (int i = 0; i < 2; ++i) {
      const int r = wm * 32 + i * 16 + l15;
      a[i].q[0] = *(const uint4*)&As[buf][r][hi * 8];
      a[i].q[1] = *(const uint4*)&As[buf][r][hi * 8 + 16];
    }
    const bf16 (*Bm[3])[40] = { Bq[buf], Bk[buf], Bv[buf] };
    #pragma unroll
    for (int m = 0; m < 3; ++m) {
      #pragma unroll
      for (int j = 0; j < 2; ++j) {
        FragU b;
        const int n = wn * 32 + j * 16 + l15;
        b.q[0] = *(const uint4*)&Bm[m][n][hi * 8];
        b.q[1] = *(const uint4*)&Bm[m][n][hi * 8 + 16];
        #pragma unroll
        for (int i = 0; i < 2; ++i)
          acc[m][i][j] = __builtin_amdgcn_wmma_f32_16x16x32_bf16(
              false, a[i].v, false, b.v, (short)0, acc[m][i][j], false, false);
      }
    }
    __syncthreads();
  }

  /* epilogue: activation, Q store, K*V / K reductions */
  #pragma unroll
  for (int j = 0; j < 2; ++j) {
    const int cn = wn * 32 + j * 16 + l15;
    const int gn = gnBase + cn;
    const float vbq = bq[gn], vbk = bk[gn], vbv = bv[gn];
    float fkv = 0.f, fks = 0.f;
    #pragma unroll
    for (int i = 0; i < 2; ++i) {
      #pragma unroll
      for (int jj = 0; jj < 8; ++jj) {
        const int gm = gmBase + wm * 32 + i * 16 + 8 * hi + jj;
        float qv = elup1(acc[0][i][j][jj] + vbq);
        float kv = elup1(acc[1][i][j][jj] + vbk);
        float vv = acc[2][i][j][jj] + vbv;
        Qb[(size_t)gm * DH + gn] = (bf16)qv;
        fkv += kv * vv;
        fks += kv;
      }
    }
    atomicAdd(&redKV[cn], fkv);
    atomicAdd(&redKS[cn], fks);
  }
  __syncthreads();
  if (tid < 64) {
    atomicAdd(&KV[(size_t)nb * DH + gnBase + tid], redKV[tid]);
    atomicAdd(&Ksum[(size_t)nb * DH + gnBase + tid], redKS[tid]);
  }
}

/* ---------------- Pass B: V' = Q*KV/(Q*Ksum+eps); out = gelu(V' @ Wo + bo) */
__global__ __launch_bounds__(256)
void out_kernel(const bf16* __restrict__ Qb,
                const float* __restrict__ KV, const float* __restrict__ Ksum,
                const bf16* __restrict__ WoT, const float* __restrict__ bo,
                float* __restrict__ out) {
  __shared__ bf16 As[2][128][40];
  __shared__ bf16 Bo[2][64][40];

  const int tid  = threadIdx.x;
  const int lane = tid & 31;
  const int w    = tid >> 5;
  const int wm   = w & 3;
  const int wn   = w >> 2;
  const int l15  = lane & 15;
  const int hi   = lane >> 4;

  const int gmBase = blockIdx.y * 128;
  const int gnBase = blockIdx.x * 64;
  const int nb     = gmBase / SEQ;

  v8f acc[2][2] = {};

  const int sc = tid & 31;   /* staging h column   */
  const int sr = tid >> 5;   /* staging row 0..7   */

  auto stage = [&](int k0, int buf) {
    /* V' tile built on the fly (nonlinear in Q -> cannot be verbatim copy) */
    const int h = k0 + sc;
    const float kvh = KV[(size_t)nb * DH + h];
    const float ksh = Ksum[(size_t)nb * DH + h];
    #pragma unroll
    for (int i = 0; i < 16; ++i) {
      const int r = sr + 8 * i;
      float q = (float)Qb[(size_t)(gmBase + r) * DH + h];
      As[buf][r][sc] = (bf16)(q * kvh / (q * ksh + 1e-6f));
    }
    /* Wo^T tile: verbatim async copy, 1 chunk per thread */
    const int n = tid >> 2, kc = (tid & 3) * 8;
    ASYNC_CP16(&Bo[buf][n][kc], WoT + (size_t)(gnBase + n) * DH + k0 + kc);
  };

  stage(0, 0);
  const int NK = DH / 32;
  for (int ks = 0; ks < NK; ++ks) {
    const int buf = ks & 1;
    if (ks + 1 < NK) { stage((ks + 1) * 32, buf ^ 1); WAIT_ASYNC(1); }
    else             { WAIT_ASYNC(0); }
    __syncthreads();

    FragU a[2];
    #pragma unroll
    for (int i = 0; i < 2; ++i) {
      const int r = wm * 32 + i * 16 + l15;
      a[i].q[0] = *(const uint4*)&As[buf][r][hi * 8];
      a[i].q[1] = *(const uint4*)&As[buf][r][hi * 8 + 16];
    }
    #pragma unroll
    for (int j = 0; j < 2; ++j) {
      FragU b;
      const int n = wn * 32 + j * 16 + l15;
      b.q[0] = *(const uint4*)&Bo[buf][n][hi * 8];
      b.q[1] = *(const uint4*)&Bo[buf][n][hi * 8 + 16];
      #pragma unroll
      for (int i = 0; i < 2; ++i)
        acc[i][j] = __builtin_amdgcn_wmma_f32_16x16x32_bf16(
            false, a[i].v, false, b.v, (short)0, acc[i][j], false, false);
    }
    __syncthreads();
  }

  #pragma unroll
  for (int j = 0; j < 2; ++j) {
    const int gn = gnBase + wn * 32 + j * 16 + l15;
    const float vbo = bo[gn];
    #pragma unroll
    for (int i = 0; i < 2; ++i) {
      #pragma unroll
      for (int jj = 0; jj < 8; ++jj) {
        const int gm = gmBase + wm * 32 + i * 16 + 8 * hi + jj;
        float xg = acc[i][j][jj] + vbo;
        float inner = 0.7978845608028654f * (xg + 0.044715f * xg * xg * xg);
        out[(size_t)gm * DH + gn] = 0.5f * xg * (1.f + tanhf(inner));
      }
    }
  }
}

extern "C" void kernel_launch(void* const* d_in, const int* in_sizes, int n_in,
                              void* d_out, int out_size, void* d_ws, size_t ws_size,
                              hipStream_t stream) {
  const float* x  = (const float*)d_in[0];
  const float* Wq = (const float*)d_in[1];
  const float* bq = (const float*)d_in[2];
  const float* Wk = (const float*)d_in[3];
  const float* bk = (const float*)d_in[4];
  const float* Wv = (const float*)d_in[5];
  const float* bv = (const float*)d_in[6];
  const float* Wo = (const float*)d_in[7];
  const float* bo = (const float*)d_in[8];

  char* ws = (char*)d_ws;
  bf16*  Qb  = (bf16*)(ws + QB_OFF);
  bf16*  xb  = (bf16*)(ws + XB_OFF);
  bf16*  WqT = (bf16*)(ws + WQT_OFF);
  bf16*  WkT = (bf16*)(ws + WKT_OFF);
  bf16*  WvT = (bf16*)(ws + WVT_OFF);
  bf16*  WoT = (bf16*)(ws + WOT_OFF);
  float* KV  = (float*)(ws + KV_OFF);
  float* Ks  = (float*)(ws + KS_OFF);

  /* x fp32 -> bf16 (verbatim layout) */
  const int xn = MROWS * DH;
  cvt_bf16_kernel<<<dim3((xn + 255) / 256), dim3(256), 0, stream>>>(x, xb, xn);

  /* weights fp32 -> bf16 TRANSPOSED (W^T[n][k]) so GEMM B-tiles are verbatim */
  dim3 tg(32, 32), tb(256);
  cvtT_kernel<<<tg, tb, 0, stream>>>(Wq, WqT);
  cvtT_kernel<<<tg, tb, 0, stream>>>(Wk, WkT);
  cvtT_kernel<<<tg, tb, 0, stream>>>(Wv, WvT);
  cvtT_kernel<<<tg, tb, 0, stream>>>(Wo, WoT);

  /* zero atomic accumulators every call (graph replays don't re-poison) */
  (void)hipMemsetAsync(ws + KV_OFF, 0, 2ull * 8 * DH * sizeof(float), stream);

  dim3 grid(DH / 64, MROWS / 128), block(256);
  qkv_kernel<<<grid, block, 0, stream>>>(xb, WqT, WkT, WvT, bq, bk, bv,
                                         Qb, KV, Ks);
  out_kernel<<<grid, block, 0, stream>>>(Qb, KV, Ks, WoT, bo, (float*)d_out);
}
